// LightModelNet_11544872091810
// MI455X (gfx1250) — matrix operand order repaired and checked by reading, not compiled, not run on hardware
//
#include <hip/hip_runtime.h>

typedef __attribute__((ext_vector_type(16))) _Float16 v16h;
typedef __attribute__((ext_vector_type(8)))  float    v8f;

#define NSITES 65536
#define NPAIR  14

// ---------------- weight packing: B-matrix VGPR layout, f16 ----------------
// B 32x16 f16 layout: element e, lanes 0-15: K=e (e<8), 16+(e-8) (e>=8), Ncol=lane;
//                              lanes 16-31: K=8+e (e<8), 24+(e-8) (e>=8), Ncol=lane&15.
// K<16 -> offset k0=2p, cin=K ; K>=16 -> offset k1=2p+1, cin=K-16.
__global__ void k_pack(const float* __restrict__ w0, const float* __restrict__ w1,
                       const float* __restrict__ w2, const float* __restrict__ w3,
                       const float* __restrict__ w4, const float* __restrict__ w5,
                       const float* __restrict__ w6, _Float16* __restrict__ bp) {
  int t = blockIdx.x * blockDim.x + threadIdx.x;
  if (t >= 7 * NPAIR * 32 * 16) return;
  int e    = t & 15;
  int lane = (t >> 4) & 31;
  int p    = (t >> 9) % NPAIR;
  int w    = t / (NPAIR * 512);
  int v = e >> 1, h = e & 1;
  int K = (e < 8) ? 8 * (lane >> 4) + e : 16 + 8 * (lane >> 4) + (e - 8);
  int k = 2 * p + (K >> 4);
  int cin = K & 15, cout = lane & 15;
  float val = 0.f;
  if (k < 27) {
    const float* W = (w == 0) ? w0 : (w == 1) ? w1 : (w == 2) ? w2 :
                     (w == 3) ? w3 : (w == 4) ? w4 : (w == 5) ? w5 : w6;
    val = W[k * 256 + cin * 16 + cout];
  }
  bp[t] = (_Float16)val;  // t == w*7168 + (p*32+lane)*16 + e
}

// ---------------- BN statistics (deterministic two-stage) ----------------
template <int CC>
__global__ void k_stats_partial(const float* __restrict__ in, float* __restrict__ part) {
  __shared__ float ssum[256], ssq[256];
  const int tid = threadIdx.x;
  const int c = tid % CC;           // compile-time modulo
  const int g = tid / CC;
  const int G = 256 / CC;
  const int base = blockIdx.x * 1024;   // 64 blocks * 1024 rows = 65536
  float sum = 0.f, sq = 0.f;
  for (int r = g; r < 1024; r += G) {
    float v = in[(size_t)(base + r) * CC + c];
    sum += v; sq += v * v;
  }
  ssum[tid] = sum; ssq[tid] = sq;
  __syncthreads();
#pragma unroll
  for (int s = G >> 1; s > 0; s >>= 1) {
    if (g < s) { ssum[tid] += ssum[tid + s * CC]; ssq[tid] += ssq[tid + s * CC]; }
    __syncthreads();
  }
  if (g == 0) {
    part[blockIdx.x * 32 + c]      = ssum[tid];
    part[blockIdx.x * 32 + 16 + c] = ssq[tid];
  }
}

template <int CC>
__global__ void k_stats_final(const float* __restrict__ part,
                              const float* __restrict__ gamma,
                              const float* __restrict__ beta,
                              float* __restrict__ aff) {
  int c = threadIdx.x;
  if (c >= 16) return;
  if (c < CC) {
    float sum = 0.f, sq = 0.f;
    for (int b = 0; b < 64; b++) { sum += part[b * 32 + c]; sq += part[b * 32 + 16 + c]; }
    float mean = sum * (1.0f / 65536.0f);
    float var  = sq  * (1.0f / 65536.0f) - mean * mean;
    float s = gamma[c] * rsqrtf(var + 1e-4f);
    aff[c] = s;
    aff[16 + c] = beta[c] - mean * s;
  } else { aff[c] = 0.f; aff[16 + c] = 0.f; }
}

// ---------------- elementwise helpers ----------------
__global__ void k_bn1(const float* __restrict__ feat, const float* __restrict__ aff,
                      float* __restrict__ y) {
  int n = blockIdx.x * blockDim.x + threadIdx.x;
  if (n >= NSITES) return;
  float v = feat[n] * aff[0] + aff[16];
  y[n] = v > 0.f ? v : 0.f;
}

__global__ void k_bn_inplace(float* __restrict__ x, const float* __restrict__ aff) {
  int i = blockIdx.x * blockDim.x + threadIdx.x;
  if (i >= NSITES * 16) return;
  int c = i & 15;
  float v = x[i] * aff[c] + aff[16 + c];
  x[i] = v > 0.f ? v : 0.f;
}

// 1->16 channel conv (tiny): scalar VALU path
__global__ void k_conv1(const float* __restrict__ y, const int* __restrict__ nbr,
                        const float* __restrict__ w, float* __restrict__ out) {
  int n = blockIdx.x * blockDim.x + threadIdx.x;
  if (n >= NSITES) return;
  float acc[16];
#pragma unroll
  for (int c = 0; c < 16; c++) acc[c] = 0.f;
  for (int k = 0; k < 27; k++) {
    int j = nbr[n * 27 + k];
    int jc = j < 0 ? 0 : j;
    float v = y[jc];
    v = (j >= 0) ? v : 0.f;
#pragma unroll
    for (int c = 0; c < 16; c++) acc[c] += v * w[k * 16 + c];
  }
#pragma unroll
  for (int c = 0; c < 16; c++) out[n * 16 + c] = acc[c];
}

// ---------------- WMMA 16->16 submanifold conv, BN-ReLU fused, residual epilogue ----
// MODE 0: out = conv ; 1: += res ; 2: += relu(res*raff) ; 3: += y[n]*Wnin[c]
template <int MODE>
__global__ void k_conv16(const float* __restrict__ xin, const int* __restrict__ nbr,
                         const _Float16* __restrict__ bp, const float* __restrict__ aff,
                         float* __restrict__ out,
                         const float* __restrict__ res, const float* __restrict__ raff,
                         const float* __restrict__ yv, const float* __restrict__ wn) {
  int lane = threadIdx.x & 31;
  int wave = threadIdx.x >> 5;
  int sb = (blockIdx.x * 8 + wave) * 16;           // 16-site tile per wave
  int m  = sb + (lane & 15);                        // this lane's gather site
  int cb = (lane < 16) ? 0 : 8;                     // channel half per ISA A-layout
  float s0[8], t0[8];
#pragma unroll
  for (int i = 0; i < 8; i++) { s0[i] = aff[cb + i]; t0[i] = aff[16 + cb + i]; }
  const v8f* x8 = (const v8f*)xin;                  // rows are 64B, halves 32B aligned
  const int hb = cb >> 3;
  v8f acc = {};
  for (int p = 0; p < NPAIR; p++) {                 // two offsets per K=32 WMMA
    int j0 = nbr[m * 27 + 2 * p];
    int j1 = (2 * p + 1 < 27) ? nbr[m * 27 + 2 * p + 1] : -1;
    // branchless gather: clamp index, load unconditionally, mask after affine
    int jc0 = j0 < 0 ? 0 : j0;
    int jc1 = j1 < 0 ? 0 : j1;
    v8f f0 = x8[(size_t)jc0 * 2 + hb];
    v8f f1 = x8[(size_t)jc1 * 2 + hb];
    v16h a;
#pragma unroll
    for (int i = 0; i < 8; i++) {
      float u = f0[i] * s0[i] + t0[i]; u = (j0 >= 0 && u > 0.f) ? u : 0.f;
      float v = f1[i] * s0[i] + t0[i]; v = (j1 >= 0 && v > 0.f) ? v : 0.f;
      a[i]     = (_Float16)u;
      a[8 + i] = (_Float16)v;
    }
    v16h b = *(const v16h*)(bp + ((p * 32 + lane) << 4));  // 32B packed B tile
    acc = __builtin_amdgcn_wmma_f32_16x16x32_f16(false, a, false, b,
                                                 (short)0, acc, false, false);
  }
  // C layout: VGPR r -> M = r (lanes 0-15) / 8+r (lanes 16-31), Ncol = lane&15
  const int cc = lane & 15;
#pragma unroll
  for (int r = 0; r < 8; r++) {
    int mr = sb + ((lane < 16) ? r : 8 + r);
    float vo = acc[r];
    if (MODE == 1)      vo += res[mr * 16 + cc];
    else if (MODE == 2) { float rr = res[mr * 16 + cc] * raff[cc] + raff[16 + cc];
                          vo += (rr > 0.f) ? rr : 0.f; }
    else if (MODE == 3)  vo += yv[mr] * wn[cc];
    out[mr * 16 + cc] = vo;
  }
}

// ---------------- sparse final linear (bn10 fused) ----------------
__global__ void k_linear_partial(const float* __restrict__ x, const float* __restrict__ aff,
                                 const long long* __restrict__ cr,
                                 const float* __restrict__ wl, float* __restrict__ part) {
  __shared__ float red[256];
  int lane = threadIdx.x & 31, wave = threadIdx.x >> 5;
  int s0 = blockIdx.x * 512 + wave * 64;            // 128 blocks = 16 per batch
  float acc = 0.f;
  for (int i = 0; i < 64; i++) {
    int s = s0 + i;
    float v = 0.f;
    if (lane < 16) {
      float u = x[s * 16 + lane] * aff[lane] + aff[16 + lane];
      v = u > 0.f ? u : 0.f;
    }
    long long cx = cr[(size_t)s * 4 + 0];
    long long cy = cr[(size_t)s * 4 + 1];
    long long cz = cr[(size_t)s * 4 + 2];
    size_t spatial = (size_t)cx * 16384 + (size_t)cy * 256 + (size_t)cz;
#pragma unroll
    for (int c = 0; c < 16; c++) {
      float vc = __shfl(v, c, 32);
      size_t row = ((size_t)c << 20) + spatial;     // ((c*D+x)*H+y)*W+z
      acc += vc * wl[row * 32 + lane];
    }
  }
  red[threadIdx.x] = acc;
  __syncthreads();
  if (threadIdx.x < 32) {
    float s = 0.f;
    for (int w = 0; w < 8; w++) s += red[w * 32 + threadIdx.x];
    part[blockIdx.x * 32 + threadIdx.x] = s;
  }
}

__global__ void k_linear_final(const float* __restrict__ part,
                               const float* __restrict__ blin, float* __restrict__ out) {
  int t = threadIdx.x;                               // 256 = 8 batches * 32 outputs
  int b = t >> 5, j = t & 31;
  float s = blin[j];
  for (int k = 0; k < 16; k++) s += part[(b * 16 + k) * 32 + j];
  out[b * 32 + j] = s;
}

// ---------------- host orchestration ----------------
extern "C" void kernel_launch(void* const* d_in, const int* in_sizes, int n_in,
                              void* d_out, int out_size, void* d_ws, size_t ws_size,
                              hipStream_t stream) {
  const long long* coord = (const long long*)d_in[0];
  const float* feat = (const float*)d_in[1];
  const int*   nbr  = (const int*)d_in[2];
  const float* W1a  = (const float*)d_in[3];
  const float* W1b  = (const float*)d_in[4];
  const float* Wnin = (const float*)d_in[5];
  const float* W2a  = (const float*)d_in[6];
  const float* W2b  = (const float*)d_in[7];
  const float* W3a  = (const float*)d_in[8];
  const float* W3b  = (const float*)d_in[9];
  const float* W4a  = (const float*)d_in[10];
  const float* W4b  = (const float*)d_in[11];
  const float* Wlin = (const float*)d_in[12];
  const float* blin = (const float*)d_in[13];
  const float* g[10]; const float* bb[10];
  for (int i = 0; i < 10; i++) { g[i] = (const float*)d_in[14 + 2 * i];
                                 bb[i] = (const float*)d_in[15 + 2 * i]; }

  char* ws = (char*)d_ws;
  float*    xbuf  = (float*)(ws);                       // 4 MB  (x)
  float*    abuf  = (float*)(ws + (4u << 20));          // 4 MB  (a)
  float*    ybuf  = (float*)(ws + (8u << 20));          // 256 KB (y scalar)
  _Float16* bpack = (_Float16*)(ws + (8u << 20) + 262144);      // 100 KB packed B
  float*    part  = (float*)(ws + (8u << 20) + 262144 + 100352); // 8 KB stats partials
  float*    aff   = part + 64 * 32;                     // 10 slots * 32 floats
  float*    lpart = aff + 10 * 32;                      // 16 KB linear partials
  float*    outf  = (float*)d_out;

  k_pack<<<196, 256, 0, stream>>>(W1b, W2a, W2b, W3a, W3b, W4a, W4b, bpack);

  auto STATS1 = [&](const float* in, int slot, const float* gg, const float* bbp) {
    k_stats_partial<1><<<64, 256, 0, stream>>>(in, part);
    k_stats_final<1><<<1, 16, 0, stream>>>(part, gg, bbp, aff + slot * 32);
  };
  auto STATS16 = [&](const float* in, int slot, const float* gg, const float* bbp) {
    k_stats_partial<16><<<64, 256, 0, stream>>>(in, part);
    k_stats_final<16><<<1, 16, 0, stream>>>(part, gg, bbp, aff + slot * 32);
  };

  // block 1, rep 0
  STATS1(feat, 0, g[0], bb[0]);
  k_bn1<<<256, 256, 0, stream>>>(feat, aff + 0, ybuf);
  k_conv1<<<256, 256, 0, stream>>>(ybuf, nbr, W1a, abuf);                 // a1
  STATS16(abuf, 1, g[1], bb[1]);
  k_conv16<3><<<512, 256, 0, stream>>>(abuf, nbr, bpack + 0 * 7168, aff + 32,
                                       xbuf, nullptr, nullptr, ybuf, Wnin); // x1 = conv+y*Wnin
  // block 1, rep 1
  STATS16(xbuf, 2, g[2], bb[2]);
  k_conv16<0><<<512, 256, 0, stream>>>(xbuf, nbr, bpack + 1 * 7168, aff + 64,
                                       abuf, nullptr, nullptr, nullptr, nullptr); // a3
  STATS16(abuf, 3, g[3], bb[3]);
  k_conv16<1><<<512, 256, 0, stream>>>(abuf, nbr, bpack + 2 * 7168, aff + 96,
                                       xbuf, xbuf, nullptr, nullptr, nullptr);    // x2 = x1+a4
  // bn5 materialized
  STATS16(xbuf, 4, g[4], bb[4]);
  k_bn_inplace<<<4096, 256, 0, stream>>>(xbuf, aff + 128);                // x3
  // block 2, rep 0
  STATS16(xbuf, 5, g[5], bb[5]);
  k_conv16<0><<<512, 256, 0, stream>>>(xbuf, nbr, bpack + 3 * 7168, aff + 160,
                                       abuf, nullptr, nullptr, nullptr, nullptr); // a5
  STATS16(abuf, 6, g[6], bb[6]);
  k_conv16<2><<<512, 256, 0, stream>>>(abuf, nbr, bpack + 4 * 7168, aff + 192,
                                       xbuf, xbuf, aff + 160, nullptr, nullptr);  // x4 = a6+relu(x3*bn6)
  // block 2, rep 1
  STATS16(xbuf, 7, g[7], bb[7]);
  k_conv16<0><<<512, 256, 0, stream>>>(xbuf, nbr, bpack + 5 * 7168, aff + 224,
                                       abuf, nullptr, nullptr, nullptr, nullptr); // a7
  STATS16(abuf, 8, g[8], bb[8]);
  k_conv16<1><<<512, 256, 0, stream>>>(abuf, nbr, bpack + 6 * 7168, aff + 256,
                                       xbuf, xbuf, nullptr, nullptr, nullptr);    // x5 = x4+a8
  // bn10 + sparse linear
  STATS16(xbuf, 9, g[9], bb[9]);
  k_linear_partial<<<128, 256, 0, stream>>>(xbuf, aff + 288, coord, Wlin, lpart);
  k_linear_final<<<1, 256, 0, stream>>>(lpart, blin, outf);
}
